// Decoder_23089744183361
// MI455X (gfx1250) — matrix operand order repaired and checked by reading, not compiled, not run on hardware
//
#include <hip/hip_runtime.h>
#include <hip/hip_bf16.h>
#include <math.h>

#define Bb   1024
#define Ss   256
#define Hh   128
#define H3   384
#define BT   16
#define NTHR 256

typedef __attribute__((ext_vector_type(16))) __bf16 v16bf;
typedef __attribute__((ext_vector_type(8)))  __bf16 v8bf;
typedef __attribute__((ext_vector_type(8)))  float  v8f;

// Shared memory: keep static LDS < 64 KB. logits aliases the dead r/z gate buffer.
struct __align__(16) Smem {
    union {
        float grz[BT][2 * Hh + 4];   // r,z pre-activations (gi+gh fused), phases 2-3
        float logits[BT][Ss + 4];    // phases 4-5
    } u;
    float   gin[BT][Hh + 4];         // i_n pre-activation
    float   ghn[BT][Hh + 4];         // h_n pre-activation
    float   h[BT][Hh + 4];           // hidden state (f32)
    __bf16  embb[BT][Hh + 8];        // emb in bf16 (A fragments), padded rows
    __bf16  hb[BT][Hh + 8];          // h in bf16 (A fragments), padded rows
    float   bih[H3];
    float   bhh[H3];
    float   bout[Ss];
    float   wemb[Hh];
    float   bemb[Hh];
    unsigned mask[BT][Ss / 32];      // visited bitset per row
    float   dec_in[BT];
};

// A-fragment (16x32 bf16, ISA 7.12.2): lane row = lane&15; lanes<16 take K={0..7,16..23},
// lanes>=16 take K={8..15,24..31} of the ks-th 32-wide K slice. Two 16B LDS loads.
__device__ __forceinline__ v16bf load_a(const __bf16* row, int ks, int half) {
    const __bf16* p0 = row + ks * 32 + half * 8;
    v8bf lo = *(const v8bf*)p0;
    v8bf hi = *(const v8bf*)(p0 + 16);
    return __builtin_shufflevector(lo, hi, 0,1,2,3,4,5,6,7,8,9,10,11,12,13,14,15);
}

// B-fragment (32x16 bf16): lane column N = j, same K order as A (K={0..7,16..23} / {8..15,24..31}).
// Since B[k][n] = W[n][k] (W row-major [N][K] = original torch layout), two 16B loads.
__device__ __forceinline__ v16bf load_b(const __bf16* __restrict__ W, int j, int ks, int half) {
    const __bf16* p0 = W + j * Hh + ks * 32 + half * 8;
    v8bf lo = *(const v8bf*)p0;
    v8bf hi = *(const v8bf*)(p0 + 16);
    return __builtin_shufflevector(lo, hi, 0,1,2,3,4,5,6,7,8,9,10,11,12,13,14,15);
}

__global__ void convert_weights(const float* __restrict__ wih, const float* __restrict__ whh,
                                const float* __restrict__ wout,
                                __bf16* __restrict__ wihb, __bf16* __restrict__ whhb,
                                __bf16* __restrict__ woutb) {
    int i = blockIdx.x * blockDim.x + threadIdx.x;
    if (i < H3 * Hh) { wihb[i] = (__bf16)wih[i]; whhb[i] = (__bf16)whh[i]; }
    if (i < Ss * Hh) { woutb[i] = (__bf16)wout[i]; }
}

__global__ void __launch_bounds__(NTHR)
decoder_kernel(const float* __restrict__ h0,
               const float* __restrict__ Wemb, const float* __restrict__ Bemb,
               const float* __restrict__ Bih,  const float* __restrict__ Bhh,
               const float* __restrict__ Bout,
               const __bf16* __restrict__ wihb, const __bf16* __restrict__ whhb,
               const __bf16* __restrict__ woutb,
               float* __restrict__ out) {
    __shared__ Smem sm;
    const int t    = threadIdx.x;
    const int wid  = t >> 5;
    const int lane = t & 31;
    const int half = lane >> 4;
    const int col  = lane & 15;
    const int bg0  = blockIdx.x * BT;

    // ---- prologue: stage h0, biases, embedding weights; clear mask ----
    for (int i = t; i < BT * Hh; i += NTHR) {
        int b = i >> 7, k = i & (Hh - 1);
        float v = h0[(bg0 + b) * Hh + k];
        sm.h[b][k]  = v;
        sm.hb[b][k] = (__bf16)v;
    }
    for (int i = t; i < H3; i += NTHR) { sm.bih[i] = Bih[i]; sm.bhh[i] = Bhh[i]; }
    for (int i = t; i < Ss; i += NTHR) sm.bout[i] = Bout[i];
    for (int i = t; i < Hh; i += NTHR) { sm.wemb[i] = Wemb[i]; sm.bemb[i] = Bemb[i]; }
    for (int i = t; i < BT * (Ss / 32); i += NTHR) (&sm.mask[0][0])[i] = 0u;
    if (t < BT) sm.dec_in[t] = 1.0f;
    __syncthreads();

    for (int step = 0; step < Ss; ++step) {
        // ---- phase 1: emb = relu(dec_in * W_emb + b_emb) -> bf16 ----
        for (int i = t; i < BT * Hh; i += NTHR) {
            int b = i >> 7, k = i & (Hh - 1);
            float e = sm.dec_in[b] * sm.wemb[k] + sm.bemb[k];
            sm.embb[b][k] = (__bf16)(e > 0.f ? e : 0.f);
        }
        __syncthreads();

        // ---- phase 2: gate GEMMs via WMMA. r/z fused (gi+gh into one accumulator) ----
        {
            v16bf Ae[4], Ah[4];
            #pragma unroll
            for (int ks = 0; ks < 4; ++ks) {
                Ae[ks] = load_a(&sm.embb[col][0], ks, half);
                Ah[ks] = load_a(&sm.hb[col][0],   ks, half);
            }
            #pragma unroll
            for (int tt = 0; tt < 2; ++tt) {             // 16 r/z tiles over 8 waves
                int n0 = 16 * (2 * wid + tt);
                v8f acc = {};
                #pragma unroll
                for (int ks = 0; ks < 4; ++ks)
                    acc = __builtin_amdgcn_wmma_f32_16x16x32_bf16(false, Ae[ks], false,
                            load_b(wihb, n0 + col, ks, half), (short)0, acc, false, false);
                #pragma unroll
                for (int ks = 0; ks < 4; ++ks)
                    acc = __builtin_amdgcn_wmma_f32_16x16x32_bf16(false, Ah[ks], false,
                            load_b(whhb, n0 + col, ks, half), (short)0, acc, false, false);
                #pragma unroll
                for (int v = 0; v < 8; ++v)
                    sm.u.grz[half * 8 + v][n0 + col] = acc[v];
            }
            {                                            // n-gate: i_n and h_n separate
                int n0 = 16 * wid;
                v8f ai = {}, ah = {};
                #pragma unroll
                for (int ks = 0; ks < 4; ++ks) {
                    ai = __builtin_amdgcn_wmma_f32_16x16x32_bf16(false, Ae[ks], false,
                            load_b(wihb, 2 * Hh + n0 + col, ks, half), (short)0, ai, false, false);
                    ah = __builtin_amdgcn_wmma_f32_16x16x32_bf16(false, Ah[ks], false,
                            load_b(whhb, 2 * Hh + n0 + col, ks, half), (short)0, ah, false, false);
                }
                #pragma unroll
                for (int v = 0; v < 8; ++v) {
                    sm.gin[half * 8 + v][n0 + col] = ai[v];
                    sm.ghn[half * 8 + v][n0 + col] = ah[v];
                }
            }
        }
        __syncthreads();

        // ---- phase 3: GRU gates + hidden update ----
        for (int i = t; i < BT * Hh; i += NTHR) {
            int b = i >> 7, k = i & (Hh - 1);
            float rv = 1.f / (1.f + __expf(-(sm.u.grz[b][k]      + sm.bih[k]      + sm.bhh[k])));
            float zv = 1.f / (1.f + __expf(-(sm.u.grz[b][Hh + k] + sm.bih[Hh + k] + sm.bhh[Hh + k])));
            float nv = tanhf(sm.gin[b][k] + sm.bih[2 * Hh + k] +
                             rv * (sm.ghn[b][k] + sm.bhh[2 * Hh + k]));
            float hn = (1.f - zv) * nv + zv * sm.h[b][k];
            sm.h[b][k]  = hn;
            sm.hb[b][k] = (__bf16)hn;
        }
        __syncthreads();

        // ---- phase 4: logits = h_new @ W_out.T + b_out (overlays grz buffer) ----
        {
            v16bf Ah[4];
            #pragma unroll
            for (int ks = 0; ks < 4; ++ks) Ah[ks] = load_a(&sm.hb[col][0], ks, half);
            #pragma unroll
            for (int tt = 0; tt < 2; ++tt) {
                int n0 = 16 * (2 * wid + tt);
                v8f acc = {};
                #pragma unroll
                for (int ks = 0; ks < 4; ++ks)
                    acc = __builtin_amdgcn_wmma_f32_16x16x32_bf16(false, Ah[ks], false,
                            load_b(woutb, n0 + col, ks, half), (short)0, acc, false, false);
                #pragma unroll
                for (int v = 0; v < 8; ++v)
                    sm.u.logits[half * 8 + v][n0 + col] = acc[v] + sm.bout[n0 + col];
            }
        }
        __syncthreads();

        // ---- phase 5: argmaxes, log-softmax, mask/dec_in update, output ----
        #pragma unroll
        for (int rr = 0; rr < 2; ++rr) {
            int b  = 2 * wid + rr;
            int c0 = lane * 8;
            unsigned mword = sm.mask[b][lane >> 2];
            float mvals[8];
            float bv = -INFINITY; int bi = 0;       // unmasked argmax (topi)
            float mv = -INFINITY; int mi = 0;       // masked argmax (chosen)
            #pragma unroll
            for (int j = 0; j < 8; ++j) {
                int c = c0 + j;
                float v = sm.u.logits[b][c];
                if (v > bv) { bv = v; bi = c; }
                float m;
                if (step == 0) m = (c == 0) ? 1.0f : -INFINITY;
                else           m = ((mword >> (c & 31)) & 1u) ? -INFINITY : v;
                mvals[j] = m;
                if (m > mv) { mv = m; mi = c; }
            }
            #pragma unroll
            for (int off = 16; off >= 1; off >>= 1) {   // butterfly: first-max tie-break
                float obv = __shfl_xor(bv, off, 32); int obi = __shfl_xor(bi, off, 32);
                if (obv > bv || (obv == bv && obi < bi)) { bv = obv; bi = obi; }
                float omv = __shfl_xor(mv, off, 32); int omi = __shfl_xor(mi, off, 32);
                if (omv > mv || (omv == mv && omi < mi)) { mv = omv; mi = omi; }
            }
            float s = 0.f;                               // logp = -log(sum exp(m - max))
            #pragma unroll
            for (int j = 0; j < 8; ++j) s += __expf(mvals[j] - mv);
            #pragma unroll
            for (int off = 16; off >= 1; off >>= 1) s += __shfl_xor(s, off, 32);
            if (lane == 0) {
                out[(bg0 + b) * Ss + step]           = (float)mi;     // tours
                out[Bb * Ss + (bg0 + b) * Ss + step] = -__logf(s);    // tour_logp
                sm.mask[b][mi >> 5] |= (1u << (mi & 31));
                sm.dec_in[b] = (float)bi;
            }
        }
        __syncthreads();
    }
}

extern "C" void kernel_launch(void* const* d_in, const int* in_sizes, int n_in,
                              void* d_out, int out_size, void* d_ws, size_t ws_size,
                              hipStream_t stream) {
    const float* h0   = (const float*)d_in[1];  // encoder_hidden [1,B,H]
    const float* wemb = (const float*)d_in[2];  // [H,1]
    const float* bemb = (const float*)d_in[3];
    const float* wih  = (const float*)d_in[4];  // [3H,H]
    const float* whh  = (const float*)d_in[5];  // [3H,H]
    const float* bih  = (const float*)d_in[6];
    const float* bhh  = (const float*)d_in[7];
    const float* wout = (const float*)d_in[8];  // [S,H]
    const float* bout = (const float*)d_in[9];

    __bf16* wihb  = (__bf16*)d_ws;
    __bf16* whhb  = wihb + H3 * Hh;
    __bf16* woutb = whhb + H3 * Hh;

    convert_weights<<<(H3 * Hh + 255) / 256, 256, 0, stream>>>(wih, whh, wout,
                                                               wihb, whhb, woutb);
    decoder_kernel<<<Bb / BT, NTHR, 0, stream>>>(h0, wemb, bemb, bih, bhh, bout,
                                                 wihb, whhb, woutb, (float*)d_out);
}
